// early_fusion_10419590660387
// MI455X (gfx1250) — matrix operand order repaired and checked by reading, not compiled, not run on hardware
//
#include <hip/hip_runtime.h>
#include <hip/hip_bf16.h>

// ---------------- problem constants ----------------
#define B_    256
#define J_    25
#define T_    64
#define MID_  12
#define IMG_  224
#define P_    (IMG_ * IMG_)       // 50176
#define OUT_  256
#define BOX_  20

typedef __attribute__((ext_vector_type(16))) _Float16 v16h;
typedef __attribute__((ext_vector_type(8)))  _Float16 h8;
typedef __attribute__((ext_vector_type(4)))  _Float16 h4;
typedef __attribute__((ext_vector_type(8)))  float    v8f;

// ---------------- k1: W fp32 -> fp16 ----------------
__global__ void k_convert_w(const float* __restrict__ W, _Float16* __restrict__ Wh, int n4) {
    int gid = blockIdx.x * blockDim.x + threadIdx.x;
    if (gid >= n4) return;
    float4 f = ((const float4*)W)[gid];
    h4 o;
    o[0] = (_Float16)f.x; o[1] = (_Float16)f.y;
    o[2] = (_Float16)f.z; o[3] = (_Float16)f.w;
    ((h4*)Wh)[gid] = o;
}

// ---------------- k2: per-(b,j) argmax frame -> box ----------------
__global__ void k_boxes(const float* __restrict__ sk, int4* __restrict__ boxes) {
    int gid = blockIdx.x * blockDim.x + threadIdx.x;   // B*J = 6400 threads
    if (gid >= B_ * J_) return;
    int b = gid / J_;
    int j = gid - b * J_;
    const float* sj = sk + ((size_t)b * J_ + j)    * T_ * 3;
    const float* sm = sk + ((size_t)b * J_ + MID_) * T_ * 3;
    float best = -1.0f; int bt = 0;
    for (int t = 0; t < T_; ++t) {
        float dx = sj[t*3+0] - sm[t*3+0];
        float dy = sj[t*3+1] - sm[t*3+1];
        float dz = sj[t*3+2] - sm[t*3+2];
        float d2 = dx*dx + dy*dy + dz*dz;
        if (d2 > best) { best = d2; bt = t; }        // strict > == first-max (jnp.argmax)
    }
    int px = (int)(sj[bt*3+0] * (float)IMG_);        // truncation, positive
    int py = (int)(sj[bt*3+1] * (float)IMG_);
    int x_lo = px - BOX_, x_hi = px + BOX_;
    int y_lo = (py < BOX_)        ? 0    : py - BOX_;
    int y_hi = (py > IMG_ - BOX_) ? IMG_ : py + BOX_;
    if (j == MID_) { x_lo = 0; x_hi = 0; }           // middle joint excluded -> empty box
    boxes[gid] = make_int4(x_lo, x_hi, y_lo, y_hi);
}

// ---------------- k3: build fp16 mask [B, 50176] ----------------
__global__ void k_mask(const int4* __restrict__ boxes, _Float16* __restrict__ mask) {
    __shared__ int4 sbox[J_];
    int b = blockIdx.x;
    if (threadIdx.x < J_) sbox[threadIdx.x] = boxes[b * J_ + threadIdx.x];
    __syncthreads();
    _Float16* mrow = mask + (size_t)b * P_;
    const int per = P_ / 256;                        // 196
    for (int i = 0; i < per; ++i) {
        int p = i * 256 + threadIdx.x;               // coalesced
        int x = p / IMG_;
        int y = p - x * IMG_;
        bool hit = false;
        for (int j = 0; j < J_; ++j) {
            int4 bb = sbox[j];
            hit = hit | ((x >= bb.x) & (x < bb.y) & (y >= bb.z) & (y < bb.w));
        }
        mrow[p] = hit ? (_Float16)1.0f : (_Float16)0.0f;
    }
}

// ---------------- k4: zero accumulator ----------------
__global__ void k_zero(float* __restrict__ acc, int n) {
    int gid = blockIdx.x * blockDim.x + threadIdx.x;
    if (gid < n) acc[gid] = 0.0f;
}

// ---------------- k5: WMMA GEMM  acc[b,o] += sum_k mask[b,k]*Wh[o,k] ----------------
// grid (8 M-tiles of 32, 4 N-tiles of 64, 8 K-splits), 256 threads = 8 waves.
#define KSPLIT 8
#define KCHUNK (P_ / KSPLIT)                         // 6272 = 196 * 32
__global__ void __launch_bounds__(256)
k_gemm(const _Float16* __restrict__ mask, const _Float16* __restrict__ Wh,
       float* __restrict__ acc) {
    const int lane = threadIdx.x & 31;
    const int wave = threadIdx.x >> 5;               // 0..7
    const int m_sub = wave & 1;                      // 2 M-subtiles
    const int n_sub = wave >> 1;                     // 4 N-subtiles

    const int m_tile = blockIdx.x * 32 + m_sub * 16;
    const int n_tile = blockIdx.y * 64 + n_sub * 16;
    const int kbase  = blockIdx.z * KCHUNK;

    // A fragment (16x32 f16): lane row = lane&15; lanes<16 carry K 0..15, lanes>=16 K 16..31
    const int a_row  = m_tile + (lane & 15);
    const int koff_a = (lane < 16) ? 0 : 16;
    // B fragment (32x16 f16): lane column = lane&15; lanes<16 carry K 0..15, lanes>=16 K 16..31
    const int b_col  = n_tile + (lane & 15);
    const int koff_b = (lane < 16) ? 0 : 16;

    const _Float16* ap = mask + (size_t)a_row * P_ + kbase + koff_a;
    const _Float16* bp = Wh   + (size_t)b_col * P_ + kbase + koff_b;

    v8f c0 = {}; v8f c1 = {};
    union AFrag { v16h v; h8 h[2]; };

    const int steps = KCHUNK / 32;                   // 196 (even)
    for (int s = 0; s < steps; s += 2) {
        AFrag a0, a1;
        a0.h[0] = *(const h8*)(ap);       a0.h[1] = *(const h8*)(ap + 8);
        a1.h[0] = *(const h8*)(ap + 32);  a1.h[1] = *(const h8*)(ap + 40);
        v16h b0 = *(const v16h*)(bp);
        v16h b1 = *(const v16h*)(bp + 32);
        c0 = __builtin_amdgcn_wmma_f32_16x16x32_f16(false, a0.v, false, b0,
                                                    (short)0, c0, false, false);
        c1 = __builtin_amdgcn_wmma_f32_16x16x32_f16(false, a1.v, false, b1,
                                                    (short)0, c1, false, false);
        ap += 64; bp += 64;
    }
    v8f c = c0 + c1;

    // D layout: VGPR r -> row m_tile + (lane<16 ? r : r+8), col n_tile + (lane&15)
    const int col = n_tile + (lane & 15);
    const int rb  = m_tile + ((lane < 16) ? 0 : 8);
    #pragma unroll
    for (int r = 0; r < 8; ++r) {
        atomicAdd(&acc[(size_t)(rb + r) * OUT_ + col], c[r]);
    }
}

// ---------------- k6: out = rgb * (acc + bias) ----------------
__global__ void k_final(const float* __restrict__ rgb, const float* __restrict__ bias,
                        const float* __restrict__ acc, float* __restrict__ out, int n) {
    int gid = blockIdx.x * blockDim.x + threadIdx.x;
    if (gid < n) out[gid] = rgb[gid] * (acc[gid] + bias[gid & (OUT_ - 1)]);
}

// ---------------- launcher ----------------
extern "C" void kernel_launch(void* const* d_in, const int* in_sizes, int n_in,
                              void* d_out, int out_size, void* d_ws, size_t ws_size,
                              hipStream_t stream) {
    const float* rgb  = (const float*)d_in[0];   // [256, 256]
    const float* sk   = (const float*)d_in[1];   // [256, 25, 64, 3]
    const float* W    = (const float*)d_in[2];   // [256, 50176]
    const float* bias = (const float*)d_in[3];   // [256]
    float* out = (float*)d_out;                  // [256, 256]

    char* ws = (char*)d_ws;
    const size_t whBytes   = (size_t)OUT_ * P_ * sizeof(_Float16);   // 25,690,112
    const size_t maskBytes = (size_t)B_   * P_ * sizeof(_Float16);   // 25,690,112
    const size_t accBytes  = (size_t)B_ * OUT_ * sizeof(float);      //    262,144
    _Float16* Wh    = (_Float16*)(ws);
    _Float16* Mask  = (_Float16*)(ws + whBytes);
    float*    Acc   = (float*)   (ws + whBytes + maskBytes);
    int4*     Boxes = (int4*)    (ws + whBytes + maskBytes + accBytes);

    // 1) W -> fp16
    {
        int n4 = OUT_ * P_ / 4;                  // 3,211,264
        k_convert_w<<<(n4 + 255) / 256, 256, 0, stream>>>(W, Wh, n4);
    }
    // 2) boxes
    k_boxes<<<(B_ * J_ + 255) / 256, 256, 0, stream>>>(sk, Boxes);
    // 3) mask
    k_mask<<<B_, 256, 0, stream>>>(Boxes, Mask);
    // 4) zero accumulator
    k_zero<<<(B_ * OUT_ + 255) / 256, 256, 0, stream>>>(Acc, B_ * OUT_);
    // 5) WMMA GEMM with K-split atomics
    {
        dim3 grid(B_ / 32, OUT_ / 64, KSPLIT);   // 8 x 4 x 8 = 256 WGs
        k_gemm<<<grid, 256, 0, stream>>>(Mask, Wh, Acc);
    }
    // 6) finalize
    k_final<<<(B_ * OUT_ + 255) / 256, 256, 0, stream>>>(rgb, bias, Acc, out, B_ * OUT_);
}
